// InvariantSRModel_51926154609210
// MI455X (gfx1250) — matrix operand order repaired and checked by reading, not compiled, not run on hardware
//
#include <hip/hip_runtime.h>
#include <math.h>

#define NSTARTS 6
#define STEPS 25
#define LRATE 0.08f
#define W6F 0.5f
#define FD_H 1e-3f
#define ADAM_B1 0.9f
#define ADAM_B2 0.999f
#define ADAM_EPS 1e-8f
#define TWO_PI_F 6.28318530717958647692f
#define PI_F 3.14159265358979323846f

typedef _Float16 v16h __attribute__((ext_vector_type(16)));
typedef float    v8f  __attribute__((ext_vector_type(8)));
typedef int      v8i  __attribute__((ext_vector_type(8)));

// ---------------------------------------------------------------------------
// Kernel 1: real SO(3) generators G_Y / G_X for l=4 and l=6 AND their squares,
// padded to 16x16, into workspace. Faithful port of _su2_generators /
// _q_real_to_complex / _so3_generators.
// Layout (floats): Y4:0  X4:256  Y6:512  X6:768  Y4^2:1024 X4^2:1280
//                  Y6^2:1536 X6^2:1792   (total 2048)
// ---------------------------------------------------------------------------
__global__ void so3_gen_kernel(float* __restrict__ wsG) {
  const int t = threadIdx.x;
  if (t >= 2) return;
  const int l = t ? 6 : 4;
  const int n = 2 * l + 1;

  float X0[13][13], Qre[13][13], Qim[13][13];
  for (int a = 0; a < 13; ++a)
    for (int b = 0; b < 13; ++b) { X0[a][b] = 0.f; Qre[a][b] = 0.f; Qim[a][b] = 0.f; }

  const float j = (float)l;
  for (int i = 0; i < n - 1; ++i) {
    float m1 = -j + (float)i;
    float rv = -sqrtf(j * (j + 1.f) - m1 * (m1 + 1.f));
    float m2 = -j + 1.f + (float)i;
    float lv =  sqrtf(j * (j + 1.f) - m2 * (m2 - 1.f));
    X0[i + 1][i] += 0.5f * rv;
    X0[i][i + 1] += 0.5f * lv;
  }

  const float is2 = 0.7071067811865475f;
  for (int m = -l; m < 0; ++m) {
    Qre[l + m][l - m] = is2;
    Qim[l + m][l + m] = -is2;
  }
  Qre[l][l] = 1.f;
  float sg = -1.f;
  for (int m = 1; m <= l; ++m) {
    Qre[l + m][l + m] = sg * is2;
    Qim[l + m][l - m] = sg * is2;
    sg = -sg;
  }
  const float ph = ((l / 2) & 1) ? -1.f : 1.f;   // (-i)^l, real for even l
  for (int a = 0; a < n; ++a)
    for (int b = 0; b < n; ++b) { Qre[a][b] *= ph; Qim[a][b] *= ph; }

  float Tre[13][13], Tim[13][13];
  for (int a = 0; a < n; ++a)
    for (int b = 0; b < n; ++b) {
      float sr = 0.f, si = 0.f;
      for (int k = 0; k < n; ++k) { sr += X0[a][k] * Qre[k][b]; si += X0[a][k] * Qim[k][b]; }
      Tre[a][b] = sr; Tim[a][b] = si;
    }

  float GY[16][16], GX[16][16];
  const int base = (l == 4) ? 0 : 512;
  for (int a = 0; a < 16; ++a)
    for (int b = 0; b < 16; ++b) {
      float gy = 0.f, gx = 0.f;
      if (a < n && b < n) {
        for (int k = 0; k < n; ++k) {
          float mk = -j + (float)k;
          gy += Qre[k][a] * (-mk * Qim[k][b]) + Qim[k][a] * (mk * Qre[k][b]);
          gx += Qre[k][a] * Tre[k][b] + Qim[k][a] * Tim[k][b];
        }
      }
      GY[a][b] = gy; GX[a][b] = gx;
      wsG[base + a * 16 + b]       = gy;
      wsG[base + 256 + a * 16 + b] = gx;
    }
  // squares (for the paired-term Taylor recurrence)
  for (int a = 0; a < 16; ++a)
    for (int b = 0; b < 16; ++b) {
      float sy = 0.f, sx = 0.f;
      for (int k = 0; k < 16; ++k) { sy += GY[a][k] * GY[k][b]; sx += GX[a][k] * GX[k][b]; }
      wsG[1024 + base + a * 16 + b]       = sy;
      wsG[1024 + base + 256 + a * 16 + b] = sx;
    }
}

// ---------------------------------------------------------------------------
// WMMA helpers (wave32). One wave carries 16 problems; lanes L and L+16 hold
// the SAME problem (column N = L%16) so EXEC stays all-ones around WMMA.
// Vector state is kept permanently in the split accumulator layout:
//   lane L holds rows r + 8*(L>=16), r = 0..7, of its column.
// A-fragment packs [G | G^2] across K=0..15 / K=16..31.
// ---------------------------------------------------------------------------

// A element e (lane L, M=L%16): lo -> G[M][e]; hi -> G2[M][e]
__device__ __forceinline__ v16h load_gen_frag2(const float* __restrict__ G,
                                               const float* __restrict__ G2, int lane) {
  const int M = lane & 15;
  const float* src = ((lane >> 4) & 1) ? G2 : G;
  v16h a;
#pragma unroll
  for (int e = 0; e < 16; ++e) {
    a[e] = (_Float16)src[M * 16 + e];
  }
  return a;
}

__device__ __forceinline__ float swapx16(float x) {
  // ds_swizzle SWAPX16: xor=0x10, and=0x1f (group-of-32)
  return __int_as_float(__builtin_amdgcn_ds_swizzle(__float_as_int(x), 0x401F));
}

// B1 fragment: lane supplies its K-half of [c1*t_full ; c1c2*t_full].
// Per lane all 16 elements use that lane's half-scale sc (c1 low / c1c2 high).
// t_full element e: e<8 -> (lo ? t[e] : o[e]); e>=8 -> (lo ? o[e-8] : t[e-8]).
__device__ __forceinline__ v16h pack_full(const float* t, const float* o,
                                          float sc, bool lo) {
  v8i bi;
#pragma unroll
  for (int i = 0; i < 4; ++i) {
    auto pt = __builtin_amdgcn_cvt_pkrtz(sc * t[2 * i], sc * t[2 * i + 1]);
    auto po = __builtin_amdgcn_cvt_pkrtz(sc * o[2 * i], sc * o[2 * i + 1]);
    int it = __builtin_bit_cast(int, pt);
    int io = __builtin_bit_cast(int, po);
    bi[i]     = lo ? it : io;
    bi[4 + i] = lo ? io : it;
  }
  return __builtin_bit_cast(v16h, bi);
}

// B2 = B1 with the K<16 half zeroed (mask2 = lo ? 0 : ~0)
__device__ __forceinline__ v16h mask_b(v16h b, int mask2) {
  v8i bi = __builtin_bit_cast(v8i, b);
#pragma unroll
  for (int i = 0; i < 8; ++i) bi[i] &= mask2;
  return __builtin_bit_cast(v16h, bi);
}

// Advance BOTH chains (va with A4=[G4|G4^2], vb with A6=[G6|G6^2]) through
// expm(theta*G): 8 substeps (phi=theta/8, ||phi*G6||<=2.4) x 10 Taylor terms,
// two terms per WMMA pair:
//   C1 = G(c1 t) + G^2(c1c2 t) = term_k + term_{k+1}   (y increment)
//   C2 = G^2(c1c2 t)           = term_{k+1}            (carried term)
// The 4 WMMAs per iteration are mutually independent -> hides the 4-coexec
// WMMA->VALU hazard and the ds_swizzle latency.
__device__ __forceinline__ void expmv_pair(v16h A4, v16h A6, float theta,
                                           float* va, float* vb,
                                           bool lo, int mask2) {
  const float phi = theta * 0.125f;
  const float phi2 = phi * phi;
  const float I1[5]  = {1.f, 1.f / 3.f, 0.2f, 1.f / 7.f, 1.f / 9.f};
  const float I12[5] = {0.5f, 1.f / 12.f, 1.f / 30.f, 1.f / 56.f, 1.f / 90.f};
#pragma unroll 1
  for (int s = 0; s < 8; ++s) {
    float ta[8], ya[8], tb[8], yb[8];
#pragma unroll
    for (int r = 0; r < 8; ++r) { ta[r] = va[r]; ya[r] = va[r]; tb[r] = vb[r]; yb[r] = vb[r]; }
#pragma unroll
    for (int i = 0; i < 5; ++i) {
      const float c1  = phi  * I1[i];
      const float c12 = phi2 * I12[i];
      const float sc  = lo ? c1 : c12;   // per-half B scale
      float oa[8], ob[8];
#pragma unroll
      for (int r = 0; r < 8; ++r) oa[r] = swapx16(ta[r]);
#pragma unroll
      for (int r = 0; r < 8; ++r) ob[r] = swapx16(tb[r]);
      v16h B1a = pack_full(ta, oa, sc, lo);
      v16h B1b = pack_full(tb, ob, sc, lo);
      v16h B2a = mask_b(B1a, mask2);
      v16h B2b = mask_b(B1b, mask2);
      const v8f Z = {0.f, 0.f, 0.f, 0.f, 0.f, 0.f, 0.f, 0.f};
      v8f C1a = __builtin_amdgcn_wmma_f32_16x16x32_f16(false, A4, false, B1a, (short)0, Z, false, false);
      v8f C1b = __builtin_amdgcn_wmma_f32_16x16x32_f16(false, A6, false, B1b, (short)0, Z, false, false);
      v8f C2a = __builtin_amdgcn_wmma_f32_16x16x32_f16(false, A4, false, B2a, (short)0, Z, false, false);
      v8f C2b = __builtin_amdgcn_wmma_f32_16x16x32_f16(false, A6, false, B2b, (short)0, Z, false, false);
#pragma unroll
      for (int r = 0; r < 8; ++r) {
        ya[r] += C1a[r]; ta[r] = C2a[r];
        yb[r] += C1b[r]; tb[r] = C2b[r];
      }
    }
#pragma unroll
    for (int r = 0; r < 8; ++r) { va[r] = ya[r]; vb[r] = yb[r]; }
  }
}

// ---------------------------------------------------------------------------
// Scalar helpers (per lane)
// ---------------------------------------------------------------------------
__device__ __forceinline__ void norm_fix(float* q) {
  float n2 = 0.f;
#pragma unroll
  for (int i = 0; i < 4; ++i) {
    float x = q[i];
    x = ((x == x) && (fabsf(x) < 1e37f)) ? x : 0.f;  // isfinite guard
    q[i] = x;
    n2 += x * x;
  }
  float n = sqrtf(n2);
  float inv = 1.f / fmaxf(n, 1e-12f);
  bool tiny = n < 1e-12f;
  q[0] = tiny ? 1.f : q[0] * inv;
  q[1] = tiny ? 0.f : q[1] * inv;
  q[2] = tiny ? 0.f : q[2] * inv;
  q[3] = tiny ? 0.f : q[3] * inv;
  float sg = (q[0] < 0.f) ? -1.f : 1.f;
#pragma unroll
  for (int i = 0; i < 4; ++i) q[i] *= sg;
}

__device__ __forceinline__ float wrap_pi(float x) {
  x = fmodf(x, TWO_PI_F);
  x = (x >  PI_F) ? x - TWO_PI_F : x;
  x = (x < -PI_F) ? x + TWO_PI_F : x;
  return x;
}

// loss(u) = mean9((D4 S4 - f4)^2) + 0.5*mean13((D6 S6 - f6)^2) with
// q = fix_sign(normalize(u)) inside. The l=4 and l=6 expm chains run in
// lockstep so their WMMAs interleave. Result identical on both lane halves.
__device__ __forceinline__ float loss_eval(const float* __restrict__ u,
                                           v16h Y4, v16h X4, v16h Y6, v16h X6,
                                           const float* __restrict__ tt4,
                                           const float* __restrict__ tt6,
                                           bool lo, int mask2) {
  float q[4] = {u[0], u[1], u[2], u[3]};
  norm_fix(q);
  const float w = q[0], x = q[1], y = q[2], z = q[3];
  const float R00 = 1.f - 2.f * (y * y + z * z), R01 = 2.f * (x * y - w * z), R02 = 2.f * (x * z + w * y);
  const float R11 = 1.f - 2.f * (x * x + z * z);
  const float R20 = 2.f * (x * z - w * y), R21 = 2.f * (y * z + w * x), R22 = 1.f - 2.f * (x * x + y * y);
  float cx = R01, cy = R11, cz = R21;
  float nn = sqrtf(cx * cx + cy * cy + cz * cz);
  float isc = 1.f / fmaxf(nn, 1e-12f);
  cx = fminf(fmaxf(cx * isc, -1.f), 1.f);
  cy = fminf(fmaxf(cy * isc, -1.f), 1.f);
  cz = fminf(fmaxf(cz * isc, -1.f), 1.f);
  float bb = acosf(cy);
  float aa = atan2f(cx, cz);
  float ca = cosf(aa), sa = sinf(aa);
  float Rp00 = ca * R00 - sa * R20;
  float Rp02 = ca * R02 - sa * R22;
  float cc = atan2f(Rp02, Rp00);
  aa = wrap_pi(aa); bb = wrap_pi(bb); cc = wrap_pi(cc);  // D is 2pi-periodic

  // split init: S4 rows {4:0.7638, 8:0.6455}; S6 rows {6:0.3536, 10:-0.9354}
  float v4[8], v6[8];
#pragma unroll
  for (int r = 0; r < 8; ++r) { v4[r] = 0.f; v6[r] = 0.f; }
  if (lo) { v4[4] = 0.7638f; v6[6] = 0.3536f; }
  else    { v4[0] = 0.6455f; v6[2] = -0.9354f; }

  expmv_pair(Y4, Y6, cc, v4, v6, lo, mask2);
  expmv_pair(X4, X6, bb, v4, v6, lo, mask2);
  expmv_pair(Y4, Y6, aa, v4, v6, lo, mask2);

  float p4 = 0.f, p6 = 0.f;
#pragma unroll
  for (int r = 0; r < 8; ++r) {
    float d4 = v4[r] - tt4[r]; p4 += d4 * d4;
    float d6 = v6[r] - tt6[r]; p6 += d6 * d6;
  }
  float l4 = (p4 + swapx16(p4)) * (1.f / 9.f);
  float l6 = (p6 + swapx16(p6)) * (1.f / 13.f);
  return l4 + W6F * l6;
}

// ---------------------------------------------------------------------------
// Kernel 2: multi-start Adam. 16 problems per wave; 4 waves per block.
// Generator tables staged into LDS with gfx1250 async global->LDS copies.
// ---------------------------------------------------------------------------
__global__ __launch_bounds__(128) void opt_kernel(
    const float* __restrict__ f4, const float* __restrict__ f6,
    const float* __restrict__ qr, const float* __restrict__ wsG,
    float* __restrict__ loss_out, float* __restrict__ q_out,
    int bsz, int nprob) {
  __shared__ float ldsG[2048];

  const int tid = threadIdx.x;
  const int lane = tid & 31;
  const int wave = blockIdx.x * (blockDim.x >> 5) + (tid >> 5);
  const bool lo = lane < 16;
  const int mask2 = lo ? 0 : ~0;     // zero the K<16 half of B2
  const int hi8 = (lane >> 4) & 1;
  const int pid0 = wave * 16 + (lane & 15);
  const int pid = (pid0 < nprob) ? pid0 : (nprob - 1);  // clamp, keep EXEC full
  const int b = pid / NSTARTS;
  const int s = pid - b * NSTARTS;

  // ---- async stage of the 8KB generator tables into LDS (ASYNCcnt path) ----
  {
    const unsigned lbase = (unsigned)(uintptr_t)&ldsG[0];
    const unsigned long long gbase = (unsigned long long)(uintptr_t)wsG;
#pragma unroll
    for (int c = 0; c < 4; ++c) {
      unsigned off = (unsigned)(tid * 16 + c * 2048);
      unsigned laddr = lbase + off;
      asm volatile("global_load_async_to_lds_b128 %0, %1, %2"
                   :: "v"(laddr), "v"(off), "s"(gbase) : "memory");
    }
    asm volatile("s_wait_asynccnt 0" ::: "memory");
    __syncthreads();
  }

  const v16h Y4 = load_gen_frag2(ldsG + 0,   ldsG + 1024, lane);
  const v16h X4 = load_gen_frag2(ldsG + 256, ldsG + 1280, lane);
  const v16h Y6 = load_gen_frag2(ldsG + 512, ldsG + 1536, lane);
  const v16h X6 = load_gen_frag2(ldsG + 768, ldsG + 1792, lane);

  __builtin_prefetch(f4 + b * 9, 0, 3);
  __builtin_prefetch(f6 + b * 13, 0, 3);

  // split targets: lane half holds rows r + 8*hi8 (out-of-range rows = 0,
  // matching the exactly-zero padded state rows)
  float tt4[8], tt6[8];
#pragma unroll
  for (int r = 0; r < 8; ++r) {
    int row = r + 8 * hi8;
    tt4[r] = (row < 9)  ? f4[b * 9 + row]  : 0.f;
    tt6[r] = (row < 13) ? f6[b * 13 + row] : 0.f;
  }

  // q0: start 0 = identity; starts 1..5 = normalize(q_rand[b][s-1])
  float u[4];
  if (s == 0) {
    u[0] = 1.f; u[1] = 0.f; u[2] = 0.f; u[3] = 0.f;
  } else {
    const float* p = qr + ((size_t)b * (NSTARTS - 1) + (s - 1)) * 4;
#pragma unroll
    for (int i = 0; i < 4; ++i) u[i] = p[i];
  }
  norm_fix(u);

  float am[4] = {0.f, 0.f, 0.f, 0.f};
  float av[4] = {0.f, 0.f, 0.f, 0.f};
  float bp1 = 1.f, bp2 = 1.f;
  const float invNK = 1.f / ((float)bsz * (float)NSTARTS);  // global-mean scale

#pragma unroll 1
  for (int step = 0; step < STEPS; ++step) {
    float g[4];
#pragma unroll 1
    for (int d = 0; d < 4; ++d) {
      float up[4] = {u[0], u[1], u[2], u[3]};
      up[d] += FD_H;
      float lp = loss_eval(up, Y4, X4, Y6, X6, tt4, tt6, lo, mask2);
      up[d] -= 2.f * FD_H;
      float lm = loss_eval(up, Y4, X4, Y6, X6, tt4, tt6, lo, mask2);
      g[d] = (lp - lm) * (0.5f / FD_H) * invNK;
    }
    bp1 *= ADAM_B1;
    bp2 *= ADAM_B2;
    const float c1 = 1.f / (1.f - bp1);
    const float c2 = 1.f / (1.f - bp2);
#pragma unroll
    for (int d = 0; d < 4; ++d) {
      am[d] = ADAM_B1 * am[d] + (1.f - ADAM_B1) * g[d];
      av[d] = ADAM_B2 * av[d] + (1.f - ADAM_B2) * g[d] * g[d];
      float mh = am[d] * c1;
      float vh = av[d] * c2;
      u[d] -= LRATE * mh / (sqrtf(vh) + ADAM_EPS);
    }
  }

  const float lf = loss_eval(u, Y4, X4, Y6, X6, tt4, tt6, lo, mask2);
  float qf[4] = {u[0], u[1], u[2], u[3]};
  norm_fix(qf);

  if (lo && pid0 < nprob) {
    loss_out[pid] = lf;
#pragma unroll
    for (int i = 0; i < 4; ++i) q_out[(size_t)pid * 4 + i] = qf[i];
  }
}

// ---------------------------------------------------------------------------
// Kernel 3: per-batch argmin over the 6 starts (first-min tie-break), final
// normalize + fix_sign into d_out.
// ---------------------------------------------------------------------------
__global__ void argmin_kernel(const float* __restrict__ loss_in,
                              const float* __restrict__ q_in,
                              float* __restrict__ out, int bsz) {
  const int b = blockIdx.x * blockDim.x + threadIdx.x;
  if (b >= bsz) return;
  int best = 0;
  float bl = loss_in[b * NSTARTS];
#pragma unroll
  for (int s = 1; s < NSTARTS; ++s) {
    float L = loss_in[b * NSTARTS + s];
    if (L < bl) { bl = L; best = s; }
  }
  float q[4];
#pragma unroll
  for (int i = 0; i < 4; ++i) q[i] = q_in[(size_t)(b * NSTARTS + best) * 4 + i];
  norm_fix(q);
#pragma unroll
  for (int i = 0; i < 4; ++i) out[(size_t)b * 4 + i] = q[i];
}

// ---------------------------------------------------------------------------
extern "C" void kernel_launch(void* const* d_in, const int* in_sizes, int n_in,
                              void* d_out, int out_size, void* d_ws, size_t ws_size,
                              hipStream_t stream) {
  const float* f4 = (const float*)d_in[0];
  const float* f6 = (const float*)d_in[1];
  const float* qr = (const float*)d_in[2];
  float* out = (float*)d_out;
  float* ws = (float*)d_ws;

  const int bsz = in_sizes[0] / 9;          // 8192
  const int nprob = bsz * NSTARTS;          // 49152

  float* wsG    = ws;                       // 2048 floats: G and G^2 tables
  float* wsLoss = ws + 2048;                // nprob floats
  float* wsQ    = ws + 2048 + nprob;        // nprob*4 floats

  so3_gen_kernel<<<1, 32, 0, stream>>>(wsG);

  const int waves = (nprob + 15) / 16;
  const int blocks = (waves + 3) / 4;       // 4 waves (128 threads) per block
  opt_kernel<<<blocks, 128, 0, stream>>>(f4, f6, qr, wsG, wsLoss, wsQ, bsz, nprob);

  argmin_kernel<<<(bsz + 255) / 256, 256, 0, stream>>>(wsLoss, wsQ, out, bsz);
}
